// MinkUNet_85126251807285
// MI455X (gfx1250) — compile-verified, matrix operand order checked by reading, not run on hardware
//
#include <hip/hip_runtime.h>

// ---------------------------------------------------------------------------
// MinkUNet forward on gfx1250: gather-GEMM via v_wmma_f32_16x16x32_bf16.
// GEMM kernel requires Ci % 32 == 0, ldi % 4 == 0, and all epilogue operands
// non-null (host substitutes ONES/ZROW/self). Odd shapes (stem Ci=3, Wdt
// Ci=16) are zero-padded on device so every call takes the one fast path.
// ---------------------------------------------------------------------------

typedef __attribute__((ext_vector_type(16))) __bf16        v16bf;
typedef __attribute__((ext_vector_type(8)))  float         v8f;
typedef __attribute__((ext_vector_type(8)))  unsigned short v8u16;

union BfPack { v16bf v; unsigned short u[16]; };

__device__ __forceinline__ unsigned short f2bf(float x) {
  union { float f; unsigned u; } c; c.f = x;
  unsigned u = c.u;
  unsigned r = u + 0x7FFFu + ((u >> 16) & 1u);   // round-to-nearest-even
  return (unsigned short)(r >> 16);
}

__device__ __forceinline__ unsigned pk2bf(float a, float b) {
  return (unsigned)f2bf(a) | ((unsigned)f2bf(b) << 16);
}

__device__ __forceinline__ float sigmoidf_(float x) { return 1.0f / (1.0f + __expf(-x)); }

// ---------------------------------------------------------------------------
// Gather-GEMM (aligned fast path, branchless epilogue):
//   O[n,co] = max( (sum_k sum_ci F[idx,ci]*W[k,ci,co]) * S[co] + Bb[co]
//                  + rmul*R[off],  floorv )
//   km == NULL -> identity gather; idx<0 -> reads Zrow (zeros).
//   Requires: Ci % 32 == 0, ldi % 4 == 0, Zrow >= Ci zero floats,
//             S/Bb/R valid pointers (R may alias O when rmul==0).
// ---------------------------------------------------------------------------
__global__ __launch_bounds__(256)
void k_gather_gemm(const float* __restrict__ F, int ldi, int Ci,
                   const float* __restrict__ W, int Co,
                   const int*   __restrict__ KM, int K,
                   const float* __restrict__ S, const float* __restrict__ Bb,
                   const float* __restrict__ R, float rmul, float floorv,
                   const float* __restrict__ Zrow,
                   float* __restrict__ O, int ldo, int cooff, int Nout)
{
  // bf16 bit-pairs stored as dwords; aliased as ushort for fragment reads.
  __shared__ __align__(16) unsigned int lA[32 * 16];   // [row][ci/2]
  __shared__ __align__(16) unsigned int lB[64 * 16];   // [co][ci/2] (transposed)
  __shared__ int sIdx[32];

  const int tid     = threadIdx.x;
  const int lane    = tid & 31;
  const int wave    = tid >> 5;
  const int rowTile = wave >> 2;        // 0..1
  const int colTile = wave & 3;         // 0..3
  const int nBase   = blockIdx.x * 32;
  const int coBase  = blockIdx.y * 64;

  const int m  = lane & 15;             // matrix row (A) / col (B) within tile
  const int kh = lane >> 4;             // K-half selector per ISA layout

  // staging assignments (fixed per thread)
  const int ar   = tid >> 3;            // A: row 0..31
  const int ac0  = (tid & 7) * 4;       // A: ci offset within 32-chunk
  const int bco  = tid & 63;            // B: co 0..63
  const int bci0 = (tid >> 6) * 8;      // B: ci run base (0,8,16,24)
  const int gco  = coBase + bco;
  const int coCl = (gco < Co) ? gco : (Co - 1);   // clamp: masked in epilogue

  const unsigned short* fragA =
      (const unsigned short*)lA + (rowTile * 16 + m) * 32 + kh * 8;
  const unsigned short* fragB =
      (const unsigned short*)lB + (colTile * 16 + m) * 32 + kh * 8;

  v8f acc = {0.f, 0.f, 0.f, 0.f, 0.f, 0.f, 0.f, 0.f};

  for (int k = 0; k < K; ++k) {
    if (tid < 32) {
      int n = nBase + tid;
      int idx = -1;
      if (n < Nout) idx = KM ? KM[(long)k * Nout + n] : n;
      sIdx[tid] = idx;
      // prefetch next tap's gather rows while this tap computes
      if (k + 1 < K && n < Nout) {
        int idx2 = KM ? KM[(long)(k + 1) * Nout + n] : n;
        if (idx2 >= 0) __builtin_prefetch(F + (long)idx2 * ldi, 0, 2);
      }
    }
    __syncthreads();

    const int    arow  = sIdx[ar];                        // ci-invariant row (per tap k)
    const float* Abase = (arow >= 0) ? (F + (long)arow * ldi) : Zrow;
    const float* Wkb   = W + (long)k * Ci * Co + coCl;

    for (int ci0 = 0; ci0 < Ci; ci0 += 32) {
      // ---- stage A: one b128 load, pack, one b64 LDS store ----
      {
        const float4 v = *(const float4*)(Abase + ci0 + ac0);
        uint2 p;
        p.x = pk2bf(v.x, v.y);
        p.y = pk2bf(v.z, v.w);
        *(uint2*)&lA[ar * 16 + (ac0 >> 1)] = p;
      }
      // ---- stage B: ci-run of 8, transposed, one b128 LDS store ----
      {
        const float* Wp = Wkb + (long)(ci0 + bci0) * Co;
        float w0 = Wp[0];        float w1 = Wp[(long)Co];
        float w2 = Wp[2L * Co];  float w3 = Wp[3L * Co];
        float w4 = Wp[4L * Co];  float w5 = Wp[5L * Co];
        float w6 = Wp[6L * Co];  float w7 = Wp[7L * Co];
        uint4 q;
        q.x = pk2bf(w0, w1);
        q.y = pk2bf(w2, w3);
        q.z = pk2bf(w4, w5);
        q.w = pk2bf(w6, w7);
        *(uint4*)&lB[bco * 16 + (bci0 >> 1)] = q;
      }
      __syncthreads();

      // ---- fragments per CDNA5 16-bit A/B layout (2x ds_load_b128 each) ----
      BfPack ap, bp;
      {
        v8u16 lo = *(const v8u16*)fragA;          // K = kh*8 .. kh*8+7
        v8u16 hi = *(const v8u16*)(fragA + 16);   // K = 16+kh*8 ..
#pragma unroll
        for (int i = 0; i < 8; ++i) { ap.u[i] = lo[i]; ap.u[i + 8] = hi[i]; }
      }
      {
        v8u16 lo = *(const v8u16*)fragB;
        v8u16 hi = *(const v8u16*)(fragB + 16);
#pragma unroll
        for (int i = 0; i < 8; ++i) { bp.u[i] = lo[i]; bp.u[i + 8] = hi[i]; }
      }

      acc = __builtin_amdgcn_wmma_f32_16x16x32_bf16(
          /*neg_a=*/false, ap.v, /*neg_b=*/false, bp.v,
          /*c_mod=*/(short)0, acc, /*reuse_a=*/false, /*reuse_b=*/false);

      __syncthreads();
    }
  }

  // ---- branchless epilogue: col = lane&15, row = r + 8*(lane>=16) ----
  const int col   = coBase + colTile * 16 + m;
  const int rbase = nBase + rowTile * 16 + kh * 8;
  if (col < Co) {
    const float sc   = S[col];
    const float bb   = Bb[col];
    const int   rmax = Nout - rbase;             // rows valid: r < rmax
    long        base = (long)rbase * ldo + cooff + col;
#pragma unroll
    for (int r = 0; r < 8; ++r) {
      if (r < rmax) {
        float v = fmaf(acc[r], sc, bb) + rmul * R[base];
        O[base] = fmaxf(v, floorv);
      }
      base += ldo;
    }
  }
}

// ---------------------------------------------------------------------------
// Padding / utility kernels
// ---------------------------------------------------------------------------
__global__ void k_fill(float* __restrict__ p, long n, float val)
{
  long t = (long)blockIdx.x * blockDim.x + threadIdx.x;
  if (t < n) p[t] = val;
}

// dst[k][ci][co] (ci padded to Cip) = ci < Ci ? src[k][ci][co] : 0
__global__ void k_pad_w(const float* __restrict__ src, int K, int Ci, int Co, int Cip,
                        float* __restrict__ dst)
{
  long t = (long)blockIdx.x * blockDim.x + threadIdx.x;
  if (t >= (long)K * Cip * Co) return;
  int co = (int)(t % Co);
  long r = t / Co;
  int ci = (int)(r % Cip);
  int k  = (int)(r / Cip);
  dst[t] = (ci < Ci) ? src[((long)k * Ci + ci) * Co + co] : 0.f;
}

// dst[n][Cp] = c < Ci ? src[n][Ci] : 0
__global__ void k_pad_rows(const float* __restrict__ src, int n, int Ci, int Cp,
                           float* __restrict__ dst)
{
  long t = (long)blockIdx.x * blockDim.x + threadIdx.x;
  if (t >= (long)n * Cp) return;
  int c = (int)(t % Cp);
  long i = t / Cp;
  dst[t] = (c < Ci) ? src[i * Ci + c] : 0.f;
}

__global__ void k_copy_cols(const float* __restrict__ src, int lds_, int cols,
                            float* __restrict__ dst, int ldd, int coff, int n)
{
  long t = (long)blockIdx.x * blockDim.x + threadIdx.x;
  if (t >= (long)n * cols) return;
  int i = (int)(t / cols), c = (int)(t % cols);
  dst[(long)i * ldd + coff + c] = src[(long)i * lds_ + c];
}

__global__ void k_add_inplace(float* __restrict__ a, const float* __restrict__ b, long n)
{
  long t = (long)blockIdx.x * blockDim.x + threadIdx.x;
  if (t < n) a[t] += b[t];
}

__global__ void k_softplus_inplace(float* __restrict__ a, long n)
{
  long t = (long)blockIdx.x * blockDim.x + threadIdx.x;
  if (t < n) {
    float v = a[t];
    a[t] = (v > 20.f) ? v : log1pf(__expf(v));
  }
}

// SE: segment mean over NBATCH=2 batches. One block of C threads.
__global__ void k_seg_mean(const float* __restrict__ f, const int* __restrict__ bi,
                           int n, int c, float* __restrict__ pool)
{
  int ch = threadIdx.x;
  float a0 = 0.f, a1 = 0.f; int c0 = 0;
  for (int i = 0; i < n; ++i) {
    float v = f[(long)i * c + ch];
    if (bi[i] == 0) { a0 += v; c0++; } else { a1 += v; }
  }
  int c1 = n - c0;
  pool[ch]     = a0 / (float)c0;
  pool[c + ch] = a1 / (float)c1;
}

// SE attention: 3-tap conv over channel axis, softmax(q_i*k_j) row-wise
__global__ void k_se_attn(const float* __restrict__ pool,
                          const float* wq, const float* bq,
                          const float* wk, const float* bk,
                          const float* wv, const float* bv,
                          float* __restrict__ ao, int C)
{
  __shared__ float kk[256], vv[256];
  int b = blockIdx.x, i = threadIdx.x;
  const float* P = pool + (long)b * C;
  float xm = (i > 0)     ? P[i - 1] : 0.f;
  float x0 = P[i];
  float xp = (i < C - 1) ? P[i + 1] : 0.f;
  float q  = wq[0] * xm + wq[1] * x0 + wq[2] * xp + bq[0];
  kk[i]    = wk[0] * xm + wk[1] * x0 + wk[2] * xp + bk[0];
  vv[i]    = wv[0] * xm + wv[1] * x0 + wv[2] * xp + bv[0];
  __syncthreads();
  float mx = -3.4e38f;
  for (int j = 0; j < C; ++j) { float e = q * kk[j]; mx = e > mx ? e : mx; }
  float se = 0.f, num = 0.f;
  for (int j = 0; j < C; ++j) {
    float e = __expf(q * kk[j] - mx);
    se += e; num += e * vv[j];
  }
  ao[(long)b * C + i] = num / se;
}

__global__ void k_se_scale(float* __restrict__ f, const float* __restrict__ ao,
                           const int* __restrict__ bi, int n, int c)
{
  long t = (long)blockIdx.x * blockDim.x + threadIdx.x;
  if (t >= (long)n * c) return;
  int i = (int)(t / c), ch = (int)(t % c);
  f[t] *= ao[(long)bi[i] * c + ch];
}

// Mamba causal depthwise conv (width 4) + SiLU
__global__ void k_mamba_conv(const float* __restrict__ xz, int ld, int di,
                             const float* __restrict__ cw, const float* __restrict__ cb_,
                             int n, float* __restrict__ xc)
{
  long t = (long)blockIdx.x * blockDim.x + threadIdx.x;
  if (t >= (long)n * di) return;
  int i = (int)(t / di), c = (int)(t % di);
  float a = 0.f;
#pragma unroll
  for (int j = 0; j < 4; ++j) {
    int r = i - 3 + j;
    float v = (r >= 0) ? xz[(long)r * ld + c] : 0.f;
    a += cw[c * 4 + j] * v;
  }
  a += cb_[c];
  xc[t] = a * sigmoidf_(a);
}

// Selective scan: thread = (d, s) with s = lane&15; sequential over tokens,
// shuffle-reduce contributions over the 16 states.
__global__ __launch_bounds__(256)
void k_scan(const float* __restrict__ dt, const float* __restrict__ A_log,
            const float* __restrict__ x, const float* __restrict__ dbl,
            int lddbl, int dtr, const float* __restrict__ Dp,
            int n, int di, float* __restrict__ yss)
{
  int s = threadIdx.x & 15;
  int d = blockIdx.x * 16 + (threadIdx.x >> 4);
  float A = -__expf(A_log[(long)d * 16 + s]);
  float Dd = Dp[d];
  float h = 0.f;
  for (int i = 0; i < n; ++i) {
    float dtv = dt[(long)i * di + d];
    float xv  = x[(long)i * di + d];
    float Bv  = dbl[(long)i * lddbl + dtr + s];
    float Cv  = dbl[(long)i * lddbl + dtr + 16 + s];
    h = __expf(dtv * A) * h + dtv * Bv * xv;
    float contrib = h * Cv;
#pragma unroll
    for (int off = 8; off > 0; off >>= 1) contrib += __shfl_xor(contrib, off, 32);
    if (s == 0) yss[(long)i * di + d] = contrib + Dd * xv;
  }
}

// Gate: ym = yss * silu(z), z = xz[:, di:]
__global__ void k_gate(const float* __restrict__ yss, const float* __restrict__ xz,
                       int ld, int di, long total, float* __restrict__ ym)
{
  long t = (long)blockIdx.x * blockDim.x + threadIdx.x;
  if (t >= total) return;
  long i = t / di, c = t % di;
  float z = xz[i * ld + di + c];
  ym[t] = yss[t] * z * sigmoidf_(z);
}

// ---------------------------------------------------------------------------
// Host orchestration
// ---------------------------------------------------------------------------
namespace {
constexpr int cN0 = 100000, cN1 = 35000, cN2 = 12000, cN3 = 4000, cN4 = 1500;
constexpr int C0 = 32, C1 = 32, C2 = 64, C3 = 128, C4 = 256;
constexpr int U1 = 256, U2 = 128, U3 = 96, U4 = 96;
constexpr int DI = 512, DTR = 16, DST = 16, NCLS = 20;
constexpr float NEGINF = -3.4e38f;

struct CB { const float* W; const float* s; const float* b; };
struct RB { CB c1, c2, down; bool has_down; };
}

extern "C" void kernel_launch(void* const* d_in, const int* in_sizes, int n_in,
                              void* d_out, int out_size, void* d_ws, size_t ws_size,
                              hipStream_t stream)
{
  (void)in_sizes; (void)n_in; (void)out_size; (void)ws_size;

  int ii = 0;
  auto nxtf = [&]() { return (const float*)d_in[ii++]; };
  auto nxti = [&]() { return (const int*)d_in[ii++]; };
  auto cb_  = [&]() { CB c; c.W = nxtf(); c.s = nxtf(); c.b = nxtf(); return c; };
  auto rb_  = [&](bool hd) { RB r; r.c1 = cb_(); r.c2 = cb_(); r.has_down = hd;
                             if (hd) r.down = cb_(); return r; };

  const float* feats = nxtf();
  const int*   bi4   = nxti();
  const int *km0 = nxti(), *km1 = nxti(), *km2 = nxti(), *km3 = nxti(), *km4 = nxti();
  const int *kd01 = nxti(), *kd12 = nxti(), *kd23 = nxti(), *kd34 = nxti();
  const int *ku43 = nxti(), *ku32 = nxti(), *ku21 = nxti(), *ku10 = nxti();

  CB stem1 = cb_(), stem2 = cb_(), down1 = cb_();
  RB s1r1 = rb_(false), s1r2 = rb_(false);
  CB down2 = cb_(); RB s2r1 = rb_(true), s2r2 = rb_(false);
  CB down3 = cb_(); RB s3r1 = rb_(true), s3r2 = rb_(false);
  CB down4 = cb_(); RB s4r1 = rb_(true), s4r2 = rb_(false);
  CB path = cb_();
  const float *wq = nxtf(), *bq = nxtf(), *wk = nxtf(), *bk = nxtf(), *wv = nxtf(), *bv = nxtf();
  const float *Win = nxtf(), *conv_w = nxtf(), *conv_b = nxtf(), *Wx = nxtf(),
              *Wdt = nxtf(), *bdt = nxtf(), *A_log = nxtf(), *Dp = nxtf(), *Wout = nxtf();
  CB up1d = cb_(); RB u1r1 = rb_(true), u1r2 = rb_(false);
  CB up2d = cb_(); RB u2r1 = rb_(true), u2r2 = rb_(false);
  CB up3d = cb_(); RB u3r1 = rb_(true), u3r2 = rb_(false);
  CB up4d = cb_(); RB u4r1 = rb_(true), u4r2 = rb_(false);
  const float* clsW = nxtf();
  const float* clsb = nxtf();

  // ---- workspace arena (floats) ----
  float* ws = (float*)d_ws;
  size_t off = 0;
  auto arena = [&](size_t n) { float* p = ws + off; off += (n + 63) & ~(size_t)63; return p; };
  float* X0 = arena((size_t)cN0 * C0);
  float* X1 = arena((size_t)cN1 * C1);
  float* X2 = arena((size_t)cN2 * C2);
  float* X3 = arena((size_t)cN3 * C3);
  float* X4 = arena((size_t)cN4 * C4);
  float* T1 = arena((size_t)cN0 * 128);
  float* T2 = arena((size_t)cN0 * 128);
  float* T3 = arena((size_t)cN0 * 96);
  float* MXZ  = arena((size_t)cN4 * 2 * DI);
  float* MXC  = arena((size_t)cN4 * DI);
  float* MDBL = arena((size_t)cN4 * (DTR + 2 * DST));
  float* MDT  = arena((size_t)cN4 * DI);
  float* MYSS = arena((size_t)cN4 * DI);
  float* MYM  = arena((size_t)cN4 * DI);
  float* MR   = arena((size_t)cN4 * C4);
  float* POOL = arena(2 * 256);
  float* AO   = arena(2 * 256);
  float* FPAD   = arena((size_t)cN0 * 32);      // feats padded 3 -> 32
  float* W1PAD  = arena((size_t)27 * 32 * C0);  // stem1 W padded ci 3 -> 32
  float* WDTPAD = arena((size_t)32 * DI);       // Wdt padded ci 16 -> 32
  float* ZROW   = arena(1536);                  // zeros (>= max Ci, >= max Co)
  float* ONES   = arena(1024);                  // ones  (>= max Co)

  auto ew = [&](long n) { return dim3((unsigned)((n + 255) / 256)); };

  // ---- one-time-per-call constants + padding (deterministic) ----
  k_fill<<<ew(1536), 256, 0, stream>>>(ZROW, 1536, 0.f);
  k_fill<<<ew(1024), 256, 0, stream>>>(ONES, 1024, 1.f);
  k_pad_rows<<<ew((long)cN0 * 32), 256, 0, stream>>>(feats, cN0, 3, 32, FPAD);
  k_pad_w<<<ew((long)27 * 32 * C0), 256, 0, stream>>>(stem1.W, 27, 3, C0, 32, W1PAD);
  k_pad_w<<<ew((long)32 * DI), 256, 0, stream>>>(Wdt, 1, DTR, DI, 32, WDTPAD);

  // ---- launch helpers (S/Bb/R substituted so kernel is branchless) ----
  auto gemm = [&](const float* F, int ldi, int Ci, const float* W, int Co,
                  const int* km, int K, const float* S, const float* Bv_,
                  const float* R, float* O, int ldo, int cooff, int Nout, int relu) {
    dim3 g((Nout + 31) / 32, (Co + 63) / 64);
    const float* Su = S   ? S   : ONES;
    const float* Bu = Bv_ ? Bv_ : ZROW;
    const float* Ru = R   ? R   : O;
    float rmul   = R ? 1.f : 0.f;
    float floorv = relu ? 0.f : NEGINF;
    k_gather_gemm<<<g, dim3(256), 0, stream>>>(F, ldi, Ci, W, Co, km, K, Su, Bu,
                                               Ru, rmul, floorv, ZROW,
                                               O, ldo, cooff, Nout);
  };
  auto cbr = [&](const float* F, int Ci, const CB& p, int Co, const int* km, int K,
                 float* O, int ldo, int cooff, int Nout) {
    gemm(F, Ci, Ci, p.W, Co, km, K, p.s, p.b, nullptr, O, ldo, cooff, Nout, 1);
  };
  auto resblock = [&](const RB& r, float* fin, int Ci, int Co, const int* km, int K,
                      int Nout, float* tmp, float* dbuf, float* out) {
    const float* res = fin;
    if (r.has_down) {
      gemm(fin, Ci, Ci, r.down.W, Co, nullptr, 1, r.down.s, r.down.b, nullptr,
           dbuf, Co, 0, Nout, 0);
      res = dbuf;
    }
    gemm(fin, Ci, Ci, r.c1.W, Co, km, K, r.c1.s, r.c1.b, nullptr, tmp, Co, 0, Nout, 1);
    gemm(tmp, Co, Co, r.c2.W, Co, km, K, r.c2.s, r.c2.b, res, out, Co, 0, Nout, 1);
  };

  // ================= encoder =================
  gemm(FPAD, 32, 32, W1PAD, C0, km0, 27, stem1.s, stem1.b, nullptr, T1, C0, 0, cN0, 1);
  cbr(T1, C0, stem2, C0, km0, 27, X0, C0, 0, cN0);

  cbr(X0, C0, down1, C0, kd01, 8, T1, C0, 0, cN1);
  resblock(s1r1, T1, C0, C1, km1, 27, cN1, T2, T3, T1);
  resblock(s1r2, T1, C1, C1, km1, 27, cN1, T2, T3, X1);

  cbr(X1, C1, down2, C1, kd12, 8, T1, C1, 0, cN2);
  resblock(s2r1, T1, C1, C2, km2, 27, cN2, T2, T3, T1);
  resblock(s2r2, T1, C2, C2, km2, 27, cN2, T2, T3, X2);

  cbr(X2, C2, down3, C2, kd23, 8, T1, C2, 0, cN3);
  resblock(s3r1, T1, C2, C3, km3, 27, cN3, T2, T3, T1);
  resblock(s3r2, T1, C3, C3, km3, 27, cN3, T2, T3, X3);

  cbr(X3, C3, down4, C3, kd34, 8, T1, C3, 0, cN4);
  resblock(s4r1, T1, C3, C4, km4, 27, cN4, T2, T3, T1);
  resblock(s4r2, T1, C4, C4, km4, 27, cN4, T2, T3, X4);

  // ================= middle: path conv + SE + Mamba =================
  cbr(X4, C4, path, C4, km4, 27, T1, C4, 0, cN4);

  k_seg_mean<<<1, 256, 0, stream>>>(T1, bi4, cN4, C4, POOL);
  k_se_attn<<<2, 256, 0, stream>>>(POOL, wq, bq, wk, bk, wv, bv, AO, C4);
  k_se_scale<<<ew((long)cN4 * C4), 256, 0, stream>>>(T1, AO, bi4, cN4, C4);

  gemm(T1, C4, C4, Win, 2 * DI, nullptr, 1, nullptr, nullptr, nullptr,
       MXZ, 2 * DI, 0, cN4, 0);
  k_mamba_conv<<<ew((long)cN4 * DI), 256, 0, stream>>>(MXZ, 2 * DI, DI, conv_w, conv_b, cN4, MXC);
  gemm(MXC, DI, DI, Wx, DTR + 2 * DST, nullptr, 1, nullptr, nullptr, nullptr,
       MDBL, DTR + 2 * DST, 0, cN4, 0);
  gemm(MDBL, DTR + 2 * DST, 32, WDTPAD, DI, nullptr, 1, nullptr, bdt, nullptr,
       MDT, DI, 0, cN4, 0);
  k_softplus_inplace<<<ew((long)cN4 * DI), 256, 0, stream>>>(MDT, (long)cN4 * DI);
  k_scan<<<dim3(DI / 16), 256, 0, stream>>>(MDT, A_log, MXC, MDBL, DTR + 2 * DST, DTR,
                                            Dp, cN4, DI, MYSS);
  k_gate<<<ew((long)cN4 * DI), 256, 0, stream>>>(MYSS, MXZ, 2 * DI, DI, (long)cN4 * DI, MYM);
  gemm(MYM, DI, DI, Wout, C4, nullptr, 1, nullptr, nullptr, nullptr, MR, C4, 0, cN4, 0);
  k_add_inplace<<<ew((long)cN4 * C4), 256, 0, stream>>>(X4, MR, (long)cN4 * C4);

  // ================= decoder =================
  // up1: N4 -> N3, concat(y[256], X3[128]) -> 384
  cbr(X4, C4, up1d, U1, ku43, 8, T1, U1 + C3, 0, cN3);
  k_copy_cols<<<ew((long)cN3 * C3), 256, 0, stream>>>(X3, C3, C3, T1, U1 + C3, U1, cN3);
  resblock(u1r1, T1, U1 + C3, U1, km3, 27, cN3, T2, T3, T1);
  resblock(u1r2, T1, U1, U1, km3, 27, cN3, T2, T3, T1);

  // up2: N3 -> N2, concat(y[128], X2[64]) -> 192
  cbr(T1, U1, up2d, U2, ku32, 8, T2, U2 + C2, 0, cN2);
  k_copy_cols<<<ew((long)cN2 * C2), 256, 0, stream>>>(X2, C2, C2, T2, U2 + C2, U2, cN2);
  resblock(u2r1, T2, U2 + C2, U2, km2, 27, cN2, T1, T3, T2);
  resblock(u2r2, T2, U2, U2, km2, 27, cN2, T1, T3, T2);

  // up3: N2 -> N1, concat(y[96], X1[32]) -> 128
  cbr(T2, U2, up3d, U3, ku21, 8, T1, U3 + C1, 0, cN1);
  k_copy_cols<<<ew((long)cN1 * C1), 256, 0, stream>>>(X1, C1, C1, T1, U3 + C1, U3, cN1);
  resblock(u3r1, T1, U3 + C1, U3, km1, 27, cN1, T2, T3, T1);
  resblock(u3r2, T1, U3, U3, km1, 27, cN1, T2, T3, T1);

  // up4: N1 -> N0, concat(y[96], X0[32]) -> 128
  cbr(T1, U3, up4d, U4, ku10, 8, T2, U4 + C0, 0, cN0);
  k_copy_cols<<<ew((long)cN0 * C0), 256, 0, stream>>>(X0, C0, C0, T2, U4 + C0, U4, cN0);
  resblock(u4r1, T2, U4 + C0, U4, km0, 27, cN0, T1, T3, T2);
  resblock(u4r2, T2, U4, U4, km0, 27, cN0, T1, T3, T2);

  // classifier
  gemm(T2, U4, U4, clsW, NCLS, nullptr, 1, nullptr, clsb, nullptr,
       (float*)d_out, NCLS, 0, cN0, 0);
}
